// AdaptiveJacobianPrunedViT_50629074485602
// MI455X (gfx1250) — matrix operand order repaired and checked by reading, not compiled, not run on hardware
//
#include <hip/hip_runtime.h>
#include <hip/hip_bf16.h>
#include <math.h>

// ---------------- model constants ----------------
#define BSZ     32
#define LAYERS  12
#define DMODEL  768
#define NHEAD   12
#define HDIM    64
#define NTOK    197           // 196 patches + CLS (max; pruning shrinks logical count)
#define NPATCH  196
#define NCLS    1000
#define RTOK    (BSZ * NTOK)  // 6304 token rows (fixed physical layout)
#define MIN_TOK 16
#define GAMMA_  0.01f
#define EPS_    1e-6f
#define SCALE_  0.125f        // 64^-0.5

// ---------------- device-side pruning state ----------------
struct VState {
    int   ntok;        // current logical token count (incl. CLS)
    int   prev_valid;
    int   prune;       // 1 if gather should run this layer
    int   nkeep;       // patches kept after pruning
    int   npatch_old;  // patch count before this layer's pruning
    float prev_mass;
};

// ---------------- bf16 helpers / WMMA types ----------------
typedef __attribute__((ext_vector_type(16))) __bf16 v16bf;
typedef __attribute__((ext_vector_type(8)))  __bf16 v8bf;
typedef __attribute__((ext_vector_type(8)))  float  v8f;

__device__ inline unsigned bfbits(float f) {
    unsigned u = __builtin_bit_cast(unsigned, f);
    return (u + 0x7fffu + ((u >> 16) & 1u)) >> 16;   // round-to-nearest-even
}
__device__ inline unsigned pack2bf(float lo, float hi) {
    return (bfbits(hi) << 16) | bfbits(lo);
}

// =====================================================================
// Generic batched GEMM: C = act(alpha*(A@B) + bias) + residual
// A [M,K] f32 -> bf16 LDS row-major; B [K,N] f32 -> bf16 LDS TRANSPOSED,
// so every WMMA fragment is a contiguous 16B-aligned LDS run (ds_load_b128).
// 128 threads = 4 waves in 2x2 tiling; each wave owns a 32x32 quadrant
// (2 A frags x 2 B frags x 4 WMMA) -> minimal redundant LDS fragment reads.
// Fast unguarded staging path for interior tiles.
// =====================================================================
#define BM 64
#define BN 64
#define BK 32
#define LPAD 8      // row pad (16 bytes) -> keeps b128 alignment, spreads banks

__global__ __launch_bounds__(128)
void gemm_wmma(const float* __restrict__ A, int lda, long long sAb, long long sAh,
               const float* __restrict__ Bm, int ldb, long long sBb, long long sBh, int transB,
               float* __restrict__ C, int ldc, long long sCb, long long sCh,
               const float* __restrict__ bias,
               const float* __restrict__ res,
               int M, int N, int K, float alpha, int act, int Hd)
{
    __shared__ __bf16 As [BM][BK + LPAD];   // row-major A tile
    __shared__ __bf16 BsT[BN][BK + LPAD];   // transposed B tile: BsT[n][k]

    const int z  = blockIdx.z;
    const int zb = z / Hd, zh = z % Hd;
    A  += (long long)zb * sAb + (long long)zh * sAh;
    Bm += (long long)zb * sBb + (long long)zh * sBh;
    C  += (long long)zb * sCb + (long long)zh * sCh;
    if (res) res += (long long)zb * sCb + (long long)zh * sCh;

    const int tid  = threadIdx.x;
    const int wave = tid >> 5;
    const int lane = tid & 31;
    const int hi   = lane >> 4;     // lane half (0/1)
    const int lr   = lane & 15;
    const int m_tile = blockIdx.y * BM;
    const int n_tile = blockIdx.x * BN;
    const int mw = (wave >> 1) * 32;   // wave quadrant within 64x64 tile
    const int nw = (wave & 1) * 32;

    v8f zero = {0.f,0.f,0.f,0.f,0.f,0.f,0.f,0.f};
    v8f acc[4] = {zero, zero, zero, zero};   // acc[ma*2+nb]

    const bool fullA = (m_tile + BM <= M);
    const bool fullB = (n_tile + BN <= N);

    for (int k0 = 0; k0 < K; k0 += BK) {
        const bool fullK = (k0 + BK <= K);

        // ---- stage A tile: pack 2 bf16 per 32-bit LDS store ----
        if (fullA && fullK) {
            #pragma unroll
            for (int i = tid; i < BM * (BK / 2); i += 128) {
                int r  = i / (BK / 2);
                int c2 = (i % (BK / 2)) * 2;
                const float* ap = A + (long long)(m_tile + r) * lda + (k0 + c2);
                *(unsigned*)(&As[r][c2]) = pack2bf(ap[0], ap[1]);
            }
        } else {
            for (int i = tid; i < BM * (BK / 2); i += 128) {
                int r  = i / (BK / 2);
                int c2 = (i % (BK / 2)) * 2;
                int gr = m_tile + r, gc = k0 + c2;
                float v0 = 0.f, v1 = 0.f;
                if (gr < M) {
                    const float* ap = A + (long long)gr * lda;
                    if (gc     < K) v0 = ap[gc];
                    if (gc + 1 < K) v1 = ap[gc + 1];
                }
                *(unsigned*)(&As[r][c2]) = pack2bf(v0, v1);
            }
        }
        // ---- stage B tile transposed: n fastest for coalesced global reads ----
        if (fullB && fullK) {
            #pragma unroll
            for (int i = tid; i < BN * (BK / 2); i += 128) {
                int n  = i % BN;
                int k2 = (i / BN) * 2;
                int gn = n_tile + n, gk = k0 + k2;
                float v0, v1;
                if (transB) {
                    const float* bp = Bm + (long long)gn * ldb + gk;
                    v0 = bp[0]; v1 = bp[1];
                } else {
                    v0 = Bm[(long long)gk * ldb + gn];
                    v1 = Bm[(long long)(gk + 1) * ldb + gn];
                }
                *(unsigned*)(&BsT[n][k2]) = pack2bf(v0, v1);
            }
        } else {
            for (int i = tid; i < BN * (BK / 2); i += 128) {
                int n  = i % BN;
                int k2 = (i / BN) * 2;
                int gn = n_tile + n, gk = k0 + k2;
                float v0 = 0.f, v1 = 0.f;
                if (gn < N) {
                    if (transB) {
                        const float* bp = Bm + (long long)gn * ldb;
                        if (gk     < K) v0 = bp[gk];
                        if (gk + 1 < K) v1 = bp[gk + 1];
                    } else {
                        if (gk     < K) v0 = Bm[(long long)gk * ldb + gn];
                        if (gk + 1 < K) v1 = Bm[(long long)(gk + 1) * ldb + gn];
                    }
                }
                *(unsigned*)(&BsT[n][k2]) = pack2bf(v0, v1);
            }
        }
        __syncthreads();

        // ---- A fragments (rows mw+ma*16+lr): two 16B runs -> 2x ds_load_b128
        // halves 0..15 -> K = hi*16 + h
        v16bf afrag[2];
        #pragma unroll
        for (int ma = 0; ma < 2; ma++) {
            const int arow = mw + ma * 16 + lr;
            v8bf a0 = *(const v8bf*)(&As[arow][hi * 16]);
            v8bf a1 = *(const v8bf*)(&As[arow][hi * 16 + 8]);
            afrag[ma] = __builtin_shufflevector(a0, a1,
                          0,1,2,3,4,5,6,7,8,9,10,11,12,13,14,15);
        }
        // ---- B fragments (cols nw+nb*16+lr): one 32B run -> 2x ds_load_b128
        v16bf bfrag[2];
        #pragma unroll
        for (int nb = 0; nb < 2; nb++) {
            const int bcol = nw + nb * 16 + lr;
            bfrag[nb] = *(const v16bf*)(&BsT[bcol][hi * 16]);
        }
        #pragma unroll
        for (int ma = 0; ma < 2; ma++)
            #pragma unroll
            for (int nb = 0; nb < 2; nb++)
                acc[ma * 2 + nb] = __builtin_amdgcn_wmma_f32_16x16x32_bf16(
                    false, afrag[ma], false, bfrag[nb],
                    (short)0, acc[ma * 2 + nb], false, false);
        __syncthreads();
    }

    // epilogue: C layout VGPR r -> M = r + hi*8, N = lane&15 (per 16x16 tile)
    #pragma unroll
    for (int ma = 0; ma < 2; ma++) {
        #pragma unroll
        for (int nb = 0; nb < 2; nb++) {
            const int col = n_tile + nw + nb * 16 + lr;
            if (col >= N) continue;
            const float bs = bias ? bias[col] : 0.0f;
            const v8f a = acc[ma * 2 + nb];
            #pragma unroll
            for (int r = 0; r < 8; r++) {
                const int row = m_tile + mw + ma * 16 + hi * 8 + r;
                if (row >= M) continue;
                float v = a[r] * alpha + bs;
                if (act == 1)  // exact GELU
                    v = 0.5f * v * (1.0f + erff(v * 0.70710678118654752f));
                const long long off = (long long)row * ldc + col;
                if (res) v += res[off];
                C[off] = v;
            }
        }
    }
}

// =====================================================================
// LayerNorm over D=768; one 256-thread block per row.
// =====================================================================
__global__ __launch_bounds__(256)
void ln_kernel(const float* __restrict__ in, long long in_rs,
               float* __restrict__ out, long long out_rs,
               const float* __restrict__ w, const float* __restrict__ bvec,
               int nrows)
{
    const int row = blockIdx.x;
    if (row >= nrows) return;
    const float* ip = in + (long long)row * in_rs;
    float* op = out + (long long)row * out_rs;
    __shared__ float red[256];

    float vals[3];
    float s = 0.f, ss = 0.f;
    #pragma unroll
    for (int i = 0; i < 3; i++) {
        float v = ip[threadIdx.x + i * 256];
        vals[i] = v; s += v; ss += v * v;
    }
    red[threadIdx.x] = s; __syncthreads();
    for (int o = 128; o > 0; o >>= 1) {
        if (threadIdx.x < o) red[threadIdx.x] += red[threadIdx.x + o];
        __syncthreads();
    }
    const float mu = red[0] * (1.0f / DMODEL);
    __syncthreads();
    red[threadIdx.x] = ss; __syncthreads();
    for (int o = 128; o > 0; o >>= 1) {
        if (threadIdx.x < o) red[threadIdx.x] += red[threadIdx.x + o];
        __syncthreads();
    }
    const float var = red[0] * (1.0f / DMODEL) - mu * mu;
    const float rs  = rsqrtf(var + EPS_);
    #pragma unroll
    for (int i = 0; i < 3; i++) {
        int d = threadIdx.x + i * 256;
        op[d] = (vals[i] - mu) * rs * w[d] + bvec[d];
    }
}

// =====================================================================
// Row softmax over logical ntok columns; zero-masks stale rows/cols.
// Also exports the CLS row (q==0) for the importance proxy.
// =====================================================================
__global__ __launch_bounds__(256)
void softmax_kernel(float* __restrict__ P, float* __restrict__ attn_cls,
                    const VState* __restrict__ st)
{
    const int z = blockIdx.y, r = blockIdx.x;
    const int nt = st->ntok;
    float* row = P + ((long long)z * NTOK + r) * NTOK;
    const int c = threadIdx.x;
    __shared__ float red[256];

    if (r >= nt) {                       // stale row: hard-zero so P@V stays clean
        if (c < NTOK) row[c] = 0.0f;
        return;
    }
    float v = (c < nt) ? row[c] * SCALE_ : -3.4e38f;
    red[c] = v; __syncthreads();
    for (int o = 128; o > 0; o >>= 1) {
        if (c < o) red[c] = fmaxf(red[c], red[c + o]);
        __syncthreads();
    }
    const float mx = red[0]; __syncthreads();
    const float e = (c < nt) ? __expf(v - mx) : 0.0f;
    red[c] = e; __syncthreads();
    for (int o = 128; o > 0; o >>= 1) {
        if (c < o) red[c] += red[c + o];
        __syncthreads();
    }
    const float inv = 1.0f / red[0];
    if (c < NTOK) {
        const float p = (c < nt) ? e * inv : 0.0f;
        row[c] = p;
        if (r == 0) attn_cls[(long long)z * NTOK + c] = p;
    }
}

// =====================================================================
// importance[j] = mean_{b,h}( attn[CLS->j+1] * ||v_{j+1}|| )
// =====================================================================
__global__ __launch_bounds__(256)
void importance_kernel(const float* __restrict__ qkv,
                       const float* __restrict__ attn_cls,
                       float* __restrict__ imp,
                       const VState* __restrict__ st)
{
    const int j  = blockIdx.x;      // patch index 0..195
    const int tj = j + 1;
    const int nt = st->ntok;
    __shared__ float red[256];
    float acc = 0.0f;
    if (tj < nt) {
        for (int z = threadIdx.x; z < BSZ * NHEAD; z += 256) {
            const int b = z / NHEAD, h = z % NHEAD;
            const float* vp = qkv + ((long long)(b * NTOK + tj)) * (3 * DMODEL)
                                  + 2 * DMODEL + h * HDIM;
            float s = 0.f;
            #pragma unroll
            for (int d = 0; d < HDIM; d++) { float x = vp[d]; s += x * x; }
            acc += attn_cls[(long long)z * NTOK + tj] * sqrtf(s);
        }
    }
    red[threadIdx.x] = acc; __syncthreads();
    for (int o = 128; o > 0; o >>= 1) {
        if (threadIdx.x < o) red[threadIdx.x] += red[threadIdx.x + o];
        __syncthreads();
    }
    if (threadIdx.x == 0)
        imp[j] = (tj < nt) ? red[0] * (1.0f / (BSZ * NHEAD)) : -1e30f;
}

// =====================================================================
// mass / rho / keep-ratio -> device-side prune decision
// =====================================================================
__global__ __launch_bounds__(256)
void stats_kernel(const float* __restrict__ imp, VState* __restrict__ st)
{
    __shared__ float red[256];
    __shared__ float sh_mass, sh_mean;
    const int np = st->ntok - 1;
    if (np <= MIN_TOK) {
        if (threadIdx.x == 0) { st->prev_valid = 0; st->prune = 0; }
        return;
    }
    float s = 0.f;
    for (int j = threadIdx.x; j < np; j += 256) s += imp[j];
    red[threadIdx.x] = s; __syncthreads();
    for (int o = 128; o > 0; o >>= 1) {
        if (threadIdx.x < o) red[threadIdx.x] += red[threadIdx.x + o];
        __syncthreads();
    }
    if (threadIdx.x == 0) { sh_mass = red[0]; sh_mean = red[0] / np; }
    __syncthreads();
    const float mean = sh_mean;
    float ss = 0.f;
    for (int j = threadIdx.x; j < np; j += 256) {
        float d = imp[j] - mean; ss += d * d;
    }
    red[threadIdx.x] = ss; __syncthreads();
    for (int o = 128; o > 0; o >>= 1) {
        if (threadIdx.x < o) red[threadIdx.x] += red[threadIdx.x + o];
        __syncthreads();
    }
    if (threadIdx.x == 0) {
        const float mass = sh_mass;
        const float stdv = sqrtf(red[0] / np);
        const float rho  = stdv / (mean + EPS_);
        int nn;
        if (st->prev_valid) {
            const float pm = st->prev_mass;
            const float drift = fabsf(mass - pm) / (pm + EPS_);
            float kr = 1.0f - GAMMA_ * rho * (1.0f + drift);
            kr = fminf(fmaxf(kr, 0.0f), 1.0f);
            nn = (int)((float)np * kr);
            if (nn < MIN_TOK) nn = MIN_TOK;
        } else {
            nn = np;
        }
        st->prev_mass = mass;
        st->prev_valid = 1;
        st->npatch_old = np;
        if (nn < np) { st->prune = 1; st->nkeep = nn; st->ntok = nn + 1; }
        else         { st->prune = 0; }
    }
}

// =====================================================================
// top-k by rank (stable, matches jax top_k + sort(idx)), builds keep_idx
// =====================================================================
__global__ __launch_bounds__(256)
void select_kernel(const float* __restrict__ imp, int* __restrict__ kidx,
                   const VState* __restrict__ st)
{
    if (!st->prune) return;
    const int np = st->npatch_old, nk = st->nkeep;
    __shared__ unsigned char keep[256];
    keep[threadIdx.x] = 0;
    __syncthreads();
    for (int j = threadIdx.x; j < np; j += 256) {
        const float me = imp[j];
        int rank = 0;
        for (int q = 0; q < np; q++) {
            const float o = imp[q];
            if (o > me || (o == me && q < j)) rank++;
        }
        keep[j] = (rank < nk) ? 1 : 0;
    }
    __syncthreads();
    for (int j = threadIdx.x; j < np; j += 256) {
        if (keep[j]) {
            int pos = 0;
            for (int q = 0; q < j; q++) pos += keep[q];
            kidx[pos + 1] = j + 1;     // +1 past CLS
        }
    }
    if (threadIdx.x == 0) kidx[0] = 0; // CLS always kept
}

__global__ __launch_bounds__(256)
void gather_kernel(const float* __restrict__ x, float* __restrict__ tmp,
                   const int* __restrict__ kidx, const VState* __restrict__ st)
{
    if (!st->prune) return;
    const int t = blockIdx.x, b = blockIdx.y;
    if (t >= st->ntok) return;
    const int src = kidx[t];
    const float* ip = x   + ((long long)(b * NTOK + src)) * DMODEL;
    float*       op = tmp + ((long long)(b * NTOK + t  )) * DMODEL;
    for (int d = threadIdx.x; d < DMODEL; d += 256) op[d] = ip[d];
}

__global__ __launch_bounds__(256)
void copyback_kernel(const float* __restrict__ tmp, float* __restrict__ x,
                     const VState* __restrict__ st)
{
    if (!st->prune) return;
    const int t = blockIdx.x, b = blockIdx.y;
    if (t >= st->ntok) return;
    const long long off = ((long long)(b * NTOK + t)) * DMODEL;
    for (int d = threadIdx.x; d < DMODEL; d += 256) x[off + d] = tmp[off + d];
}

// =====================================================================
// patch embedding: im2col + assemble (GEMM does conv as [B*196,768]x[768,768])
// =====================================================================
__global__ __launch_bounds__(256)
void im2col_kernel(const float* __restrict__ x, float* __restrict__ out)
{
    const long long total = (long long)BSZ * NPATCH * DMODEL;
    long long idx = (long long)blockIdx.x * 256 + threadIdx.x;
    if (idx >= total) return;
    const int col = (int)(idx % DMODEL);
    const long long row = idx / DMODEL;
    const int b = (int)(row / NPATCH), p = (int)(row % NPATCH);
    const int py = p / 14, px = p % 14;
    const int c  = col >> 8, rem = col & 255;
    const int ky = rem >> 4, kx = rem & 15;
    out[idx] = x[(((long long)(b * 3 + c) * 224) + py * 16 + ky) * 224
                 + px * 16 + kx];
}

__global__ __launch_bounds__(256)
void assemble_kernel(const float* __restrict__ feat, const float* __restrict__ cls,
                     const float* __restrict__ pos, float* __restrict__ x)
{
    const long long total = (long long)RTOK * DMODEL;
    long long idx = (long long)blockIdx.x * 256 + threadIdx.x;
    if (idx >= total) return;
    const int d = (int)(idx % DMODEL);
    const long long r = idx / DMODEL;
    const int t = (int)(r % NTOK), b = (int)(r / NTOK);
    const float v = (t == 0) ? cls[d]
                             : feat[((long long)(b * NPATCH + t - 1)) * DMODEL + d];
    x[idx] = v + pos[(long long)t * DMODEL + d];
}

__global__ void init_state_kernel(VState* st)
{
    st->ntok = NTOK;
    st->prev_valid = 0;
    st->prune = 0;
    st->nkeep = NPATCH;
    st->npatch_old = NPATCH;
    st->prev_mass = 0.0f;
}

// =====================================================================
// host-side orchestration
// =====================================================================
extern "C" void kernel_launch(void* const* d_in, const int* in_sizes, int n_in,
                              void* d_out, int out_size, void* d_ws, size_t ws_size,
                              hipStream_t stream)
{
    (void)in_sizes; (void)n_in; (void)out_size; (void)ws_size;

    const float* xin     = (const float*)d_in[0];
    const float* patch_w = (const float*)d_in[1];   // [768, 768] (N-major -> transB)
    const float* patch_b = (const float*)d_in[2];
    const float* cls_t   = (const float*)d_in[3];
    const float* pos_e   = (const float*)d_in[4];
    const float* ln1_w   = (const float*)d_in[5];
    const float* ln1_b   = (const float*)d_in[6];
    const float* qkv_w   = (const float*)d_in[7];
    const float* qkv_b   = (const float*)d_in[8];
    const float* proj_w  = (const float*)d_in[9];
    const float* proj_b  = (const float*)d_in[10];
    const float* ln2_w   = (const float*)d_in[11];
    const float* ln2_b   = (const float*)d_in[12];
    const float* fc1_w   = (const float*)d_in[13];
    const float* fc1_b   = (const float*)d_in[14];
    const float* fc2_w   = (const float*)d_in[15];
    const float* fc2_b   = (const float*)d_in[16];
    const float* norm_w  = (const float*)d_in[17];
    const float* norm_b  = (const float*)d_in[18];
    const float* head_w  = (const float*)d_in[19];
    const float* head_b  = (const float*)d_in[20];
    float* out = (float*)d_out;

    // workspace layout (floats)
    VState* st = (VState*)d_ws;
    float* base = (float*)d_ws + 64;
    float* x        = base;                                   // [RTOK, 768]
    float* xn       = x        + (long long)RTOK * DMODEL;    // [RTOK, 768] (LN out / im2col / gather tmp)
    float* qkv      = xn       + (long long)RTOK * DMODEL;    // [RTOK, 2304]
    float* big      = qkv      + (long long)RTOK * 3 * DMODEL;// [RTOK, 3072] (scores / fc1 / feat)
    float* attn_out = big      + (long long)RTOK * 4 * DMODEL;// [RTOK, 768]
    float* attn_cls = attn_out + (long long)RTOK * DMODEL;    // [384, 197]
    float* imp      = attn_cls + (long long)BSZ * NHEAD * NTOK; // [256]
    int*   kidx     = (int*)(imp + 256);                      // [256]

    auto gemm = [&](const float* A, int lda, long long sAb, long long sAh,
                    const float* Bp, int ldb, long long sBb, long long sBh, int tB,
                    float* Cp, int ldc, long long sCb, long long sCh,
                    const float* bias, const float* res,
                    int M, int N, int K, float alpha, int act, int Z, int Hd) {
        dim3 g((N + BN - 1) / BN, (M + BM - 1) / BM, Z);
        gemm_wmma<<<g, dim3(128), 0, stream>>>(A, lda, sAb, sAh,
                                               Bp, ldb, sBb, sBh, tB,
                                               Cp, ldc, sCb, sCh,
                                               bias, res, M, N, K, alpha, act, Hd);
    };

    init_state_kernel<<<1, 1, 0, stream>>>(st);

    // ---- patch embedding: im2col -> GEMM(conv) -> assemble(cls+pos) ----
    {
        const long long tot = (long long)BSZ * NPATCH * DMODEL;
        im2col_kernel<<<(unsigned)((tot + 255) / 256), 256, 0, stream>>>(xin, xn);
        gemm(xn, DMODEL, 0, 0,
             patch_w, DMODEL, 0, 0, /*transB=*/1,
             big, DMODEL, 0, 0,
             patch_b, nullptr,
             BSZ * NPATCH, DMODEL, DMODEL, 1.0f, 0, 1, 1);
        const long long tot2 = (long long)RTOK * DMODEL;
        assemble_kernel<<<(unsigned)((tot2 + 255) / 256), 256, 0, stream>>>(big, cls_t, pos_e, x);
    }

    const long long tokQKV = (long long)NTOK * 3 * DMODEL;   // qkv batch stride per image
    const long long sS     = (long long)NTOK * NTOK;         // scores per (b,h)

    for (int i = 0; i < LAYERS; i++) {
        const float* l1w = ln1_w + (long long)i * DMODEL;
        const float* l1b = ln1_b + (long long)i * DMODEL;
        const float* l2w = ln2_w + (long long)i * DMODEL;
        const float* l2b = ln2_b + (long long)i * DMODEL;
        const float* qw  = qkv_w + (long long)i * DMODEL * 3 * DMODEL;
        const float* qb  = qkv_b + (long long)i * 3 * DMODEL;
        const float* pw  = proj_w + (long long)i * DMODEL * DMODEL;
        const float* pb  = proj_b + (long long)i * DMODEL;
        const float* f1w = fc1_w + (long long)i * DMODEL * 4 * DMODEL;
        const float* f1b = fc1_b + (long long)i * 4 * DMODEL;
        const float* f2w = fc2_w + (long long)i * 4 * DMODEL * DMODEL;
        const float* f2b = fc2_b + (long long)i * DMODEL;

        // LN1
        ln_kernel<<<RTOK, 256, 0, stream>>>(x, DMODEL, xn, DMODEL, l1w, l1b, RTOK);
        // QKV GEMM: [6304,768] x [768,2304]
        gemm(xn, DMODEL, 0, 0, qw, 3 * DMODEL, 0, 0, 0,
             qkv, 3 * DMODEL, 0, 0, qb, nullptr,
             RTOK, 3 * DMODEL, DMODEL, 1.0f, 0, 1, 1);
        // scores = Q @ K^T per (b,h): batched 384
        gemm(qkv, 3 * DMODEL, tokQKV, HDIM,
             qkv + DMODEL, 3 * DMODEL, tokQKV, HDIM, /*transB=*/1,
             big, NTOK, (long long)NHEAD * sS, sS,
             nullptr, nullptr,
             NTOK, NTOK, HDIM, 1.0f, 0, BSZ * NHEAD, NHEAD);
        // softmax (applies SCALE, masks to logical ntok, exports CLS row)
        softmax_kernel<<<dim3(NTOK, BSZ * NHEAD), 256, 0, stream>>>(big, attn_cls, st);
        // out = P @ V, written directly into [B, N, H*64] layout
        gemm(big, NTOK, (long long)NHEAD * sS, sS,
             qkv + 2 * DMODEL, 3 * DMODEL, tokQKV, HDIM, 0,
             attn_out, DMODEL, (long long)NTOK * DMODEL, HDIM,
             nullptr, nullptr,
             NTOK, HDIM, NTOK, 1.0f, 0, BSZ * NHEAD, NHEAD);
        // proj + residual: x = x + attn_out @ proj_w + b
        gemm(attn_out, DMODEL, 0, 0, pw, DMODEL, 0, 0, 0,
             x, DMODEL, 0, 0, pb, x,
             RTOK, DMODEL, DMODEL, 1.0f, 0, 1, 1);
        // LN2
        ln_kernel<<<RTOK, 256, 0, stream>>>(x, DMODEL, xn, DMODEL, l2w, l2b, RTOK);
        // FC1 + exact GELU
        gemm(xn, DMODEL, 0, 0, f1w, 4 * DMODEL, 0, 0, 0,
             big, 4 * DMODEL, 0, 0, f1b, nullptr,
             RTOK, 4 * DMODEL, DMODEL, 1.0f, /*gelu=*/1, 1, 1);
        // FC2 + residual
        gemm(big, 4 * DMODEL, 0, 0, f2w, DMODEL, 0, 0, 0,
             x, DMODEL, 0, 0, f2b, x,
             RTOK, DMODEL, 4 * DMODEL, 1.0f, 0, 1, 1);

        // ---- adaptive pruning (fully device-side; no-ops when prune==0) ----
        importance_kernel<<<NPATCH, 256, 0, stream>>>(qkv, attn_cls, imp, st);
        stats_kernel<<<1, 256, 0, stream>>>(imp, st);
        select_kernel<<<1, 256, 0, stream>>>(imp, kidx, st);
        gather_kernel<<<dim3(NTOK, BSZ), 256, 0, stream>>>(x, xn, kidx, st);
        copyback_kernel<<<dim3(NTOK, BSZ), 256, 0, stream>>>(xn, x, st);
    }

    // final LN on CLS rows only, then head GEMM
    ln_kernel<<<BSZ, 256, 0, stream>>>(x, (long long)NTOK * DMODEL,
                                       xn, DMODEL, norm_w, norm_b, BSZ);
    gemm(xn, DMODEL, 0, 0, head_w, NCLS, 0, 0, 0,
         out, NCLS, 0, 0, head_b, nullptr,
         BSZ, NCLS, DMODEL, 1.0f, 0, 1, 1);
}
